// RNNDecoder_9698036155131
// MI455X (gfx1250) — compile-verified, matrix operand order checked by reading, not run on hardware
//
#include <hip/hip_runtime.h>
#include <stddef.h>

// Problem constants (from the reference)
#define Bsz 16
#define Nn  64
#define Dd  4
#define Tt  40
#define Ll  10
#define Hh  128
#define Kk  4
#define Ee  (Nn * (Nn - 1))   // 4032
#define MT  8                 // 16-row M tiles per msg block (128 edge-rows)
#define MROWS (16 * MT)

// ---------------------------------------------------------------------------
// WMMA fragment types (CDNA5 / gfx1250, wave32)
// ---------------------------------------------------------------------------
typedef __bf16          v16bf     __attribute__((ext_vector_type(16)));
typedef float           v8f       __attribute__((ext_vector_type(8)));
typedef unsigned short  ushort16v __attribute__((ext_vector_type(16)));
typedef unsigned short  ushort8v  __attribute__((ext_vector_type(8)));
typedef unsigned short  ushort4v  __attribute__((ext_vector_type(4)));

union FragU {
    v16bf     bf;
    ushort16v u;
    struct Halves { ushort8v lo, hi; } h;
};

__device__ __forceinline__ unsigned short f2bf(float f) {
    unsigned int u = __float_as_uint(f);
    u += 0x7FFFu + ((u >> 16) & 1u);   // round-to-nearest-even
    return (unsigned short)(u >> 16);
}

// gfx1250 hardware transcendental tanh (v_tanh_f32); exp-based fallback.
__device__ __forceinline__ float fast_tanh(float x) {
#if __has_builtin(__builtin_amdgcn_tanhf)
    return __builtin_amdgcn_tanhf(x);
#else
    float e = __expf(-2.0f * x);
    return (1.0f - e) / (1.0f + e);
#endif
}
__device__ __forceinline__ float fast_sigmoid(float x) {
    return 0.5f * fast_tanh(0.5f * x) + 0.5f;
}

// Async global->LDS copy (CDNA5): 16 bytes per lane, no VGPR staging,
// tracked by ASYNCcnt. LDS destination address is a 32-bit byte offset
// (low 32 bits of the flat shared-aperture pointer, per ISA 10.2).
__device__ __forceinline__ void async_copy16(unsigned lds_addr, const void* gaddr) {
    asm volatile("global_load_async_to_lds_b128 %0, %1, off"
                 :: "v"(lds_addr), "v"(gaddr)
                 : "memory");
}
__device__ __forceinline__ void wait_async0() {
#if __has_builtin(__builtin_amdgcn_s_wait_asynccnt)
    __builtin_amdgcn_s_wait_asynccnt(0);
#else
    asm volatile("s_wait_asynccnt 0x0" ::: "memory");
#endif
}

// 16-bit A matrix 16x32 layout (ISA 7.12.2): lane&15 = M row, lane>>4 picks the
// K-half; elements 0..7 = K[half*8..+7], elements 8..15 = K[16+half*8..+7].
__device__ __forceinline__ v16bf lds_load_a(const unsigned short* lds, int ldk, int k0) {
    const int lane = threadIdx.x & 31;
    const unsigned short* p = lds + (lane & 15) * ldk + k0 + (lane >> 4) * 8;
    FragU f;
    f.h.lo = *(const ushort8v*)(p);
    f.h.hi = *(const ushort8v*)(p + 16);
    return f.bf;
}

// B fragments pre-swizzled in global memory: 16 contiguous bf16 (32 B) per lane.
__device__ __forceinline__ v16bf gbl_load_b(const unsigned short* packed, int fragIdx) {
    FragU f;
    f.u = ((const ushort16v*)packed)[fragIdx];
    return f.bf;
}

__device__ __forceinline__ v8f wmma_bf16(v16bf a, v16bf b, v8f c) {
    return __builtin_amdgcn_wmma_f32_16x16x32_bf16(
        false, a, false, b, (short)0, c, false, false);
}

// ---------------------------------------------------------------------------
// Weight packing: src fp32 [Kdim][128] row-major -> bf16 B-fragment stream.
// ---------------------------------------------------------------------------
__global__ void pack_weight_kernel(const float* __restrict__ src,
                                   unsigned short* __restrict__ dst, int Kdim) {
    const int KT = Kdim >> 5;
    const int total = Kdim * Hh;
    for (int p = blockIdx.x * blockDim.x + threadIdx.x; p < total;
         p += gridDim.x * blockDim.x) {
        int j    = p & 15;
        int lane = (p >> 4) & 31;
        int kt   = (p >> 9) % KT;
        int nt   = (p >> 9) / KT;
        int half = lane >> 4;
        int n    = nt * 16 + (lane & 15);
        int kin  = (j < 8) ? (half * 8 + j) : (16 + half * 8 + (j - 8));
        dst[p] = f2bf(src[(size_t)(kt * 32 + kin) * Hh + n]);
    }
}

__global__ void init_state_kernel(float* hidden, float* agg, float* prev,
                                  unsigned short* hiddenBf) {
    int i = blockIdx.x * blockDim.x + threadIdx.x;
    if (i < Bsz * Nn * Hh) { hidden[i] = 0.0f; agg[i] = 0.0f; hiddenBf[i] = 0; }
    if (i < Bsz * Nn * Dd) prev[i] = 0.0f;
}

// ---------------------------------------------------------------------------
// Edge message-pass kernel: one block = 128 edge-rows (8 M-tiles of 16), 8
// waves each owning one 16-wide N tile. Each B fragment is loaded ONCE and
// reused across all 8 M-tiles. Per wave: 4 edge types x (8+4)x8 = 384 WMMAs.
// preMsg is gathered straight into LDS with async b128 copies (bf16 mirror of
// hidden), so staging needs no VGPR round-trip and no conversion VALU.
// Dynamic LDS: preMsg 64KB + m1 32KB + relt 2KB + rows 0.5KB ~= 101KB.
// ---------------------------------------------------------------------------
__global__ __launch_bounds__(256) void msg_pass_kernel(
    const unsigned short* __restrict__ hiddenBf, const float* __restrict__ rel_type,
    const unsigned short* __restrict__ pw1, const float* __restrict__ b1,
    const unsigned short* __restrict__ pw2, const float* __restrict__ b2,
    const int* __restrict__ recv_idx, const int* __restrict__ send_idx,
    float* __restrict__ agg) {

    extern __shared__ unsigned char smem[];
    unsigned short* preMsg  = (unsigned short*)smem;              // [MROWS][2*Hh]
    unsigned short* m1S     = preMsg + MROWS * 2 * Hh;            // [MROWS][Hh]
    float*          reltS   = (float*)(m1S + MROWS * Hh);         // [MROWS][Kk]
    int*            aggRowS = (int*)(reltS + MROWS * Kk);         // [MROWS]

    const int tid  = threadIdx.x;
    const int lane = tid & 31;
    const int nt   = tid >> 5;                 // wave id == N column tile
    const int r0   = blockIdx.x * MROWS;       // global row in [0, B*E)

    // per-row metadata: rel_type (float4) + receiver scatter base
    for (int lr = tid; lr < MROWS; lr += 256) {
        int r = r0 + lr;
        int b = r / Ee;
        int e = r - b * Ee;
        aggRowS[lr] = (b * Nn + recv_idx[e]) * Hh;
        *(float4*)&reltS[lr * Kk] = *(const float4*)&rel_type[(size_t)r * Kk];
    }
    // preMsg[lr][:] = concat(hiddenBf[b, send[e]], hiddenBf[b, recv[e]])
    // as async 16B-per-lane global->LDS copies (32 chunks of 16B per row).
    {
        const unsigned preMsgOff = (unsigned)(uintptr_t)preMsg;
        for (int q = tid; q < MROWS * 32; q += 256) {
            int lr = q >> 5;
            int ch = q & 31;                       // 16B chunk within row
            int r  = r0 + lr;
            int b  = r / Ee;
            int e  = r - b * Ee;
            int node = (ch < 16) ? send_idx[e] : recv_idx[e];
            const unsigned short* gsrc =
                hiddenBf + ((size_t)b * Nn + node) * Hh + (ch & 15) * 8;
            unsigned lds_addr = preMsgOff + (unsigned)(lr * (2 * Hh) + ch * 8) * 2;
            async_copy16(lds_addr, (const void*)gsrc);
        }
        wait_async0();
    }
    __syncthreads();

    const int half = lane >> 4;
    const int ncol = nt * 16 + (lane & 15);
    const v8f vzero = {};
    v8f accAll[MT];
    #pragma unroll
    for (int mt = 0; mt < MT; ++mt) accAll[mt] = vzero;

    for (int k = 0; k < Kk; ++k) {
        // ---- layer 1: [128 x 256] @ [256 x 128] ----
        const unsigned short* w1k = pw1 + (size_t)k * (2 * Hh * Hh);
        v8f c1[MT];
        #pragma unroll
        for (int mt = 0; mt < MT; ++mt) c1[mt] = vzero;
        for (int kt = 0; kt < 8; ++kt) {
            v16bf bfrag = gbl_load_b(w1k, (nt * 8 + kt) * 32 + lane);
            if (kt + 1 < 8)
                __builtin_prefetch(w1k + ((size_t)((nt * 8 + kt + 1) * 32 + lane)) * 16, 0, 1);
            #pragma unroll
            for (int mt = 0; mt < MT; ++mt) {
                v16bf a = lds_load_a(preMsg + mt * 16 * (2 * Hh), 2 * Hh, kt * 32);
                c1[mt] = wmma_bf16(a, bfrag, c1[mt]);
            }
        }
        const float bias1 = b1[k * Hh + ncol];
        #pragma unroll
        for (int mt = 0; mt < MT; ++mt)
            #pragma unroll
            for (int v = 0; v < 8; ++v)
                m1S[(mt * 16 + v + 8 * half) * Hh + ncol] =
                    f2bf(fast_tanh(c1[mt][v] + bias1));
        __syncthreads();

        // ---- layer 2: [128 x 128] @ [128 x 128] ----
        const unsigned short* w2k = pw2 + (size_t)k * (Hh * Hh);
        v8f c2[MT];
        #pragma unroll
        for (int mt = 0; mt < MT; ++mt) c2[mt] = vzero;
        for (int kt = 0; kt < 4; ++kt) {
            v16bf bfrag = gbl_load_b(w2k, (nt * 4 + kt) * 32 + lane);
            if (kt + 1 < 4)
                __builtin_prefetch(w2k + ((size_t)((nt * 4 + kt + 1) * 32 + lane)) * 16, 0, 1);
            #pragma unroll
            for (int mt = 0; mt < MT; ++mt) {
                v16bf a = lds_load_a(m1S + mt * 16 * Hh, Hh, kt * 32);
                c2[mt] = wmma_bf16(a, bfrag, c2[mt]);
            }
        }
        const float bias2 = b2[k * Hh + ncol];
        #pragma unroll
        for (int mt = 0; mt < MT; ++mt)
            #pragma unroll
            for (int v = 0; v < 8; ++v) {
                int row = mt * 16 + v + 8 * half;
                accAll[mt][v] += reltS[row * Kk + k] * fast_tanh(c2[mt][v] + bias2);
            }
        __syncthreads();   // m1S reused next edge type
    }

    // scatter-add to receivers; fold /K (edge-type avg) and /D (n_in_node)
    const float scale = 1.0f / (float)(Kk * Dd);
    #pragma unroll
    for (int mt = 0; mt < MT; ++mt)
        #pragma unroll
        for (int v = 0; v < 8; ++v) {
            int row = mt * 16 + v + 8 * half;
            atomicAdd(&agg[(size_t)aggRowS[row] + ncol], accAll[mt][v] * scale);
        }
}

// ---------------------------------------------------------------------------
// Node update kernel: one block = 16 (b,n) rows. GRU + output MLP, closed-loop
// input selection, output write for the last L steps, agg clear, and the bf16
// hidden mirror for the next step's async message gather.
// ---------------------------------------------------------------------------
__global__ __launch_bounds__(256) void node_update_kernel(
    const float* __restrict__ data,
    const unsigned short* __restrict__ pwr, const unsigned short* __restrict__ pwi,
    const unsigned short* __restrict__ pwh,
    const float* __restrict__ Wr_in, const float* __restrict__ br_in,
    const float* __restrict__ Wi_in, const float* __restrict__ bi_in,
    const float* __restrict__ Wn_in, const float* __restrict__ bn_in,
    const unsigned short* __restrict__ po1, const float* __restrict__ bo1,
    const unsigned short* __restrict__ po2, const float* __restrict__ bo2,
    const float* __restrict__ Wo3, const float* __restrict__ bo3,
    float* __restrict__ hidden, unsigned short* __restrict__ hiddenBf,
    float* __restrict__ agg, float* __restrict__ prev,
    float* __restrict__ out, int step) {

    __shared__ unsigned short aggS[16][Hh];
    __shared__ unsigned short hidS[16][Hh];
    __shared__ unsigned short y1S[16][Hh];
    __shared__ float y2S[16][Hh];
    __shared__ float insS[16][Dd];

    const int tid  = threadIdx.x;
    const int lane = tid & 31;
    const int nt   = tid >> 5;
    const int r0   = blockIdx.x * 16;   // flat (b*N + n) row base

    if (tid < 64) {
        int m = tid >> 2, d = tid & 3;
        int row = r0 + m;
        float x = (step < Tt) ? data[((size_t)row * Dd + d) * Tt + step]
                              : prev[(size_t)row * Dd + d];
        insS[m][d] = x;
    }
    for (int idx = tid; idx < 16 * Hh / 4; idx += 256) {
        int m = idx >> 5, c4 = (idx & 31) * 4;
        size_t g = ((size_t)(r0 + m)) * Hh + c4;
        float4 v = *(const float4*)&agg[g];
        ushort4v o = { f2bf(v.x), f2bf(v.y), f2bf(v.z), f2bf(v.w) };
        *(ushort4v*)&aggS[m][c4] = o;
        float4 z = { 0.f, 0.f, 0.f, 0.f };
        *(float4*)&agg[g] = z;          // reset for next step's scatter-add
    }
    __syncthreads();

    // GRU recurrent GEMMs on agg: r/i/h linear terms (12 WMMA per wave)
    v8f cR = {}, cI = {}, cN = {};
    for (int kt = 0; kt < 4; ++kt) {
        v16bf a = lds_load_a(&aggS[0][0], Hh, kt * 32);
        cR = wmma_bf16(a, gbl_load_b(pwr, (nt * 4 + kt) * 32 + lane), cR);
        cI = wmma_bf16(a, gbl_load_b(pwi, (nt * 4 + kt) * 32 + lane), cI);
        cN = wmma_bf16(a, gbl_load_b(pwh, (nt * 4 + kt) * 32 + lane), cN);
    }
    const int half = lane >> 4;
    const int h    = nt * 16 + (lane & 15);

    // tiny D(=4)xH input matmuls as per-lane scalar FMAs (column h per lane)
    float wr[Dd], wi[Dd], wn[Dd];
    #pragma unroll
    for (int d = 0; d < Dd; ++d) {
        wr[d] = Wr_in[d * Hh + h];
        wi[d] = Wi_in[d * Hh + h];
        wn[d] = Wn_in[d * Hh + h];
    }
    const float brv = br_in[h], biv = bi_in[h], bnv = bn_in[h];

    #pragma unroll
    for (int v = 0; v < 8; ++v) {
        int m = v + 8 * half;
        int row = r0 + m;
        float xr = brv, xi = biv, xn = bnv;
        #pragma unroll
        for (int d = 0; d < Dd; ++d) {
            float x = insS[m][d];
            xr = fmaf(x, wr[d], xr);
            xi = fmaf(x, wi[d], xi);
            xn = fmaf(x, wn[d], xn);
        }
        float r = fast_sigmoid(xr + cR[v]);
        float i = fast_sigmoid(xi + cI[v]);
        float n = fast_tanh(xn + r * cN[v]);
        size_t g = (size_t)row * Hh + h;
        float hold = hidden[g];
        float hnew = (1.0f - i) * n + i * hold;
        hidden[g] = hnew;
        unsigned short hb = f2bf(hnew);
        hiddenBf[g] = hb;               // bf16 mirror for async msg gather
        hidS[m][h] = hb;
    }
    __syncthreads();

    // y1 = relu(hidden @ Wo1 + bo1)
    v8f c1 = {};
    for (int kt = 0; kt < 4; ++kt) {
        v16bf a = lds_load_a(&hidS[0][0], Hh, kt * 32);
        c1 = wmma_bf16(a, gbl_load_b(po1, (nt * 4 + kt) * 32 + lane), c1);
    }
    {
        const float bv = bo1[h];
        #pragma unroll
        for (int v = 0; v < 8; ++v) {
            float y = c1[v] + bv;
            y1S[v + 8 * half][h] = f2bf(y > 0.0f ? y : 0.0f);
        }
    }
    __syncthreads();

    // y2 = relu(y1 @ Wo2 + bo2) -> fp32 LDS for the final HxD projection
    v8f c2 = {};
    for (int kt = 0; kt < 4; ++kt) {
        v16bf a = lds_load_a(&y1S[0][0], Hh, kt * 32);
        c2 = wmma_bf16(a, gbl_load_b(po2, (nt * 4 + kt) * 32 + lane), c2);
    }
    {
        const float bv = bo2[h];
        #pragma unroll
        for (int v = 0; v < 8; ++v) {
            float y = c2[v] + bv;
            y2S[v + 8 * half][h] = (y > 0.0f ? y : 0.0f);
        }
    }
    __syncthreads();

    // pred = ins + y2 @ Wo3 + bo3 ; feed back + emit last-L outputs
    if (tid < 64) {
        int m = tid >> 2, d = tid & 3;
        float s = bo3[d];
        for (int c = 0; c < Hh; ++c) s = fmaf(y2S[m][c], Wo3[c * Dd + d], s);
        float pred = insS[m][d] + s;
        int row = r0 + m;
        prev[(size_t)row * Dd + d] = pred;
        if (step >= Tt)
            out[((size_t)row * Dd + d) * Ll + (step - Tt)] = pred;
    }
}

// ---------------------------------------------------------------------------
// Host launcher
// ---------------------------------------------------------------------------
extern "C" void kernel_launch(void* const* d_in, const int* in_sizes, int n_in,
                              void* d_out, int out_size, void* d_ws, size_t ws_size,
                              hipStream_t stream) {
    (void)in_sizes; (void)n_in; (void)out_size; (void)ws_size;

    const float* data     = (const float*)d_in[0];
    const float* rel_type = (const float*)d_in[1];
    const float* msg_W1   = (const float*)d_in[2];
    const float* msg_b1   = (const float*)d_in[3];
    const float* msg_W2   = (const float*)d_in[4];
    const float* msg_b2   = (const float*)d_in[5];
    const float* Wr_h     = (const float*)d_in[6];
    const float* Wi_h     = (const float*)d_in[7];
    const float* Wh_h     = (const float*)d_in[8];
    const float* Wr_in    = (const float*)d_in[9];
    const float* br_in    = (const float*)d_in[10];
    const float* Wi_in    = (const float*)d_in[11];
    const float* bi_in    = (const float*)d_in[12];
    const float* Wn_in    = (const float*)d_in[13];
    const float* bn_in    = (const float*)d_in[14];
    const float* Wo1      = (const float*)d_in[15];
    const float* bo1      = (const float*)d_in[16];
    const float* Wo2      = (const float*)d_in[17];
    const float* bo2      = (const float*)d_in[18];
    const float* Wo3      = (const float*)d_in[19];
    const float* bo3      = (const float*)d_in[20];
    const int* recv_idx   = (const int*)d_in[21];
    const int* send_idx   = (const int*)d_in[22];

    char* ws = (char*)d_ws;
    size_t off = 0;
    auto carve = [&](size_t bytes) -> void* {
        off = (off + 255) & ~(size_t)255;
        void* p = ws + off;
        off += bytes;
        return p;
    };
    float* hidden = (float*)carve((size_t)Bsz * Nn * Hh * sizeof(float));
    float* agg    = (float*)carve((size_t)Bsz * Nn * Hh * sizeof(float));
    float* prev   = (float*)carve((size_t)Bsz * Nn * Dd * sizeof(float));
    unsigned short* hiddenBf = (unsigned short*)carve((size_t)Bsz * Nn * Hh * 2);
    unsigned short* pw1 = (unsigned short*)carve((size_t)Kk * 2 * Hh * Hh * 2);
    unsigned short* pw2 = (unsigned short*)carve((size_t)Kk * Hh * Hh * 2);
    unsigned short* pwr = (unsigned short*)carve((size_t)Hh * Hh * 2);
    unsigned short* pwi = (unsigned short*)carve((size_t)Hh * Hh * 2);
    unsigned short* pwh = (unsigned short*)carve((size_t)Hh * Hh * 2);
    unsigned short* po1 = (unsigned short*)carve((size_t)Hh * Hh * 2);
    unsigned short* po2 = (unsigned short*)carve((size_t)Hh * Hh * 2);

    for (int k = 0; k < Kk; ++k) {
        pack_weight_kernel<<<(2 * Hh * Hh) / 256, 256, 0, stream>>>(
            msg_W1 + (size_t)k * 2 * Hh * Hh, pw1 + (size_t)k * 2 * Hh * Hh, 2 * Hh);
        pack_weight_kernel<<<(Hh * Hh) / 256, 256, 0, stream>>>(
            msg_W2 + (size_t)k * Hh * Hh, pw2 + (size_t)k * Hh * Hh, Hh);
    }
    pack_weight_kernel<<<(Hh * Hh) / 256, 256, 0, stream>>>(Wr_h, pwr, Hh);
    pack_weight_kernel<<<(Hh * Hh) / 256, 256, 0, stream>>>(Wi_h, pwi, Hh);
    pack_weight_kernel<<<(Hh * Hh) / 256, 256, 0, stream>>>(Wh_h, pwh, Hh);
    pack_weight_kernel<<<(Hh * Hh) / 256, 256, 0, stream>>>(Wo1, po1, Hh);
    pack_weight_kernel<<<(Hh * Hh) / 256, 256, 0, stream>>>(Wo2, po2, Hh);

    init_state_kernel<<<(Bsz * Nn * Hh + 255) / 256, 256, 0, stream>>>(
        hidden, agg, prev, hiddenBf);

    // dynamic LDS for the msg kernel: preMsg + m1 + relt + scatter rows
    const size_t msg_smem =
        (size_t)MROWS * 2 * Hh * 2 +   // preMsg bf16
        (size_t)MROWS * Hh * 2 +       // m1 bf16
        (size_t)MROWS * Kk * 4 +       // rel_type
        (size_t)MROWS * 4;             // aggRow bases

    for (int step = 0; step < Tt + Ll; ++step) {
        msg_pass_kernel<<<(Bsz * Ee) / MROWS, 256, msg_smem, stream>>>(
            hiddenBf, rel_type, pw1, msg_b1, pw2, msg_b2, recv_idx, send_idx, agg);
        node_update_kernel<<<(Bsz * Nn) / 16, 256, 0, stream>>>(
            data, pwr, pwi, pwh, Wr_in, br_in, Wi_in, bi_in, Wn_in, bn_in,
            po1, bo1, po2, bo2, Wo3, bo3, hidden, hiddenBf, agg, prev,
            (float*)d_out, step);
    }
}
